// F0Resonance_67568425501575
// MI455X (gfx1250) — matrix operand order repaired and checked by reading, not compiled, not run on hardware
//
#include <hip/hip_runtime.h>
#include <math.h>

typedef __attribute__((ext_vector_type(16))) _Float16 v16h;
typedef __attribute__((ext_vector_type(8)))  float    v8f;

#define N_OCT   16
#define N_SMP   32768
#define CHUNK   2048
#define TILES   (CHUNK / 256)     // 8 tiles of 16x16 samples per wave-chunk
#define CHUNKS  (N_SMP / CHUNK)   // 16 chunks per row
#define MINF    (20.0f  / 11025.0f)
#define MAXF    (3000.0f / 11025.0f)
#define PI_F    3.14159265358979323846f

// ---------------- kernel 1: per-row params (f_o, w_o) + zero max ----------------
__global__ void k_params(const float* __restrict__ f0in,
                         const float* __restrict__ dcin,
                         const float* __restrict__ fsin,
                         float* __restrict__ ws_f,
                         float* __restrict__ ws_w,
                         unsigned* __restrict__ maxbits,
                         int rows)
{
    int r = blockIdx.x * blockDim.x + threadIdx.x;
    if (r >= rows) return;
    maxbits[r] = 0u;

    float f0v = fabsf(f0in[r]);
    f0v = (MINF + f0v * (MAXF - MINF)) * PI_F;

    // double sigmoid (reference applies sigmoid twice)
    float s1 = 1.0f / (1.0f + expf(-dcin[r]));
    float s2 = 1.0f / (1.0f + expf(-s1));
    float decay = 0.01f + s2 * (0.99f * 0.99f) + 1e-12f;  // + eps as in log(decay+1e-12)

    float spacing = fsin[r];
    float wcur = 1.0f;
    #pragma unroll
    for (int o = 0; o < N_OCT; ++o) {
        float fo = f0v * spacing * (float)(o + 1);
        if (!(fo < 1.0f)) fo = 0.0f;          // nyquist cutoff: f0s * (f0s < 1.0)
        wcur *= decay;                        // decay^(o+1)
        ws_f[r * N_OCT + o] = fo;
        ws_w[r * N_OCT + o] = wcur;
    }
}

// ---------------- kernel 2: WMMA oscillator-bank synthesis ----------------
// One wave per (row, 2048-sample chunk). Per 256-sample tile:
//   D = A(16x32 f16) x B(32x16 f16)  via v_wmma_f32_16x16x32_f16
//   A[m][k<16]  = w_k * sin(f_k * P_m),  A[m][k>=16] = w_{k-16} * cos(f_{k-16} * P_m)
//   B[k<16][n]  = cos(f_k * (n+1)),      B[k>=16][n] = sin(f_{k-16} * (n+1))
//   => D[m][n] = sum_o w_o * sin(f_o * (s+1)),  s = tile_base + 16m + n
// The decay weight w is folded into the rotation state (rotation preserves scale),
// so the A fragment is a pure f32->f16 convert of the state each tile.
__global__ void __launch_bounds__(256)
k_synth(const float* __restrict__ ws_f,
        const float* __restrict__ ws_w,
        unsigned* __restrict__ maxbits,
        float* __restrict__ out)
{
    const int lane = threadIdx.x & 31;
    const int wave = threadIdx.x >> 5;
    const int wgid = blockIdx.x * 8 + wave;     // global wave id
    const int row  = wgid >> 4;                 // wgid / CHUNKS
    const int chnk = wgid & (CHUNKS - 1);
    const int hi   = lane >> 4;                 // lane half
    const int mn   = lane & 15;                 // M for A-side / N for B,D-side

    const float* f = ws_f + row * N_OCT;
    const float* w = ws_w + row * N_OCT;

    // B fragment (constant for the whole row): h<8 -> K = hi*8 + h (cos), h>=8 -> K = 16 + hi*8 + h-8 (sin), octave o = hi*8 + (h&7)
    v16h bfrag;
    #pragma unroll
    for (int h = 0; h < N_OCT; ++h) {
        float ang = f[hi * 8 + (h & 7)] * (float)(mn + 1);
        float v = (h >= 8) ? sinf(ang) : cosf(ang);
        bfrag[h] = (_Float16)v;
    }

    // A-side rotation state for this lane's 8 octaves (o = hi*8 + j), M = mn.
    // State carries the decay weight: sn = w*sin(phase), cs = w*cos(phase).
    float sn[8], cs[8], c2[8], s2[8];
    const float P0 = (float)(chnk * CHUNK + 16 * mn);
    #pragma unroll
    for (int j = 0; j < 8; ++j) {
        float fo = f[hi * 8 + j];
        float wj = w[hi * 8 + j];
        float a0 = fo * P0;
        sn[j] = wj * sinf(a0);  cs[j] = wj * cosf(a0);
        float a1 = fo * 256.0f;                  // phase advance per tile
        s2[j] = sinf(a1);       c2[j] = cosf(a1);
    }

    float mx = 0.0f;
    float* orow = out + (size_t)row * N_SMP + chnk * CHUNK;

    #pragma unroll
    for (int t = 0; t < TILES; ++t) {
        // A fragment: half h<8 -> weighted sin (K = hi*8+h), h>=8 -> weighted cos (K = 16+hi*8+h-8)
        v16h a;
        #pragma unroll
        for (int j = 0; j < 8; ++j) {
            a[j]     = (_Float16)sn[j];
            a[j + 8] = (_Float16)cs[j];
        }
        v8f d = {};
        d = __builtin_amdgcn_wmma_f32_16x16x32_f16(
                /*neg_a=*/false, a, /*neg_b=*/false, bfrag,
                /*c_mod=*/(short)0, d, /*reuse_a=*/false, /*reuse_b=*/false);

        float* ot = orow + t * 256;
        #pragma unroll
        for (int r2 = 0; r2 < 8; ++r2) {        // D: M = r2 + hi*8, N = mn
            float v = d[r2];
            ot[16 * (r2 + hi * 8) + mn] = v;
            mx = fmaxf(mx, fabsf(v));
        }
        // rotate (weighted) phases forward by 256 samples
        #pragma unroll
        for (int j = 0; j < 8; ++j) {
            float ns = sn[j] * c2[j] + cs[j] * s2[j];
            cs[j]    = cs[j] * c2[j] - sn[j] * s2[j];
            sn[j]    = ns;
        }
    }

    // wave-level max reduction, then one atomicMax per wave (|x| >= 0 -> uint order == float order)
    #pragma unroll
    for (int off = 16; off > 0; off >>= 1)
        mx = fmaxf(mx, __shfl_xor(mx, off, 32));
    if (lane == 0)
        atomicMax(maxbits + row, __float_as_uint(mx));
}

// ---------------- kernel 3: in-place max-norm ----------------
__global__ void k_norm(float* __restrict__ out,
                       const unsigned* __restrict__ maxbits,
                       int n4)
{
    int i = blockIdx.x * blockDim.x + threadIdx.x;
    if (i >= n4) return;
    int row = i >> 13;                            // (i*4) / 32768
    float m = __uint_as_float(maxbits[row]);
    float inv = 1.0f / (m + 1e-8f);
    float4* o4 = (float4*)out;
    float4 v = o4[i];
    v.x *= inv; v.y *= inv; v.z *= inv; v.w *= inv;
    o4[i] = v;
}

extern "C" void kernel_launch(void* const* d_in, const int* in_sizes, int n_in,
                              void* d_out, int out_size, void* d_ws, size_t ws_size,
                              hipStream_t stream)
{
    const float* f0 = (const float*)d_in[0];
    const float* dc = (const float*)d_in[1];
    const float* fs = (const float*)d_in[3];
    float* out = (float*)d_out;

    const int rows = in_sizes[0];               // B*E = 512

    unsigned* maxbits = (unsigned*)d_ws;
    float* ws_f = (float*)d_ws + rows;
    float* ws_w = ws_f + rows * N_OCT;

    k_params<<<(rows + 255) / 256, 256, 0, stream>>>(f0, dc, fs, ws_f, ws_w, maxbits, rows);

    const int waves  = rows * CHUNKS;           // 8192 waves
    k_synth<<<waves / 8, 256, 0, stream>>>(ws_f, ws_w, maxbits, out);

    const int n4 = out_size / 4;
    k_norm<<<(n4 + 255) / 256, 256, 0, stream>>>(out, maxbits, n4);
}
